// Vocoder_40003325395707
// MI455X (gfx1250) — matrix operand order repaired and checked by reading, not compiled, not run on hardware
//
#include <hip/hip_runtime.h>
#include <hip/hip_bf16.h>

// ---------------------------------------------------------------------------
// Types for CDNA5 WMMA (wave32): v_wmma_f32_16x16x32_bf16
// ---------------------------------------------------------------------------
typedef __attribute__((ext_vector_type(16))) __bf16 v16bf;
typedef __attribute__((ext_vector_type(8)))  __bf16 v8bf;
typedef __attribute__((ext_vector_type(8)))  float  v8f;

__device__ inline v8f wmma_bf16(v16bf a, v16bf b, v8f c) {
  // (neg_a, A, neg_b, B, c_mod, C, reuse_a, reuse_b)
  return __builtin_amdgcn_wmma_f32_16x16x32_bf16(false, a, false, b,
                                                 (short)0, c, false, false);
}

__device__ inline v8f zero_v8f() {
  v8f z;
#pragma unroll
  for (int i = 0; i < 8; ++i) z[i] = 0.0f;
  return z;
}

__device__ inline float sigmoidf_(float x) { return 1.0f / (1.0f + __expf(-x)); }

// ---------------------------------------------------------------------------
// CDNA5 async memory->LDS copy (gfx1250): per-lane 16B transfer, ASYNCcnt.
// LDS byte offset = low 32 bits of the generic (shared-aperture) address.
// ---------------------------------------------------------------------------
__device__ inline unsigned lds_off(const void* p) {
  return (unsigned)(unsigned long long)p;
}
__device__ inline void async_ld_b128(unsigned lds, const void* gaddr) {
  asm volatile("global_load_async_to_lds_b128 %0, %1, off"
               :: "v"(lds), "v"(gaddr) : "memory");
}
__device__ inline void wait_async0() {
  asm volatile("s_wait_asynccnt 0x0" ::: "memory");
}
// Steady-state pipeline wait: tile i landed, tile i+1 (4 transfers) in flight.
__device__ inline void wait_async4() {
  asm volatile("s_wait_asynccnt 0x4" ::: "memory");
}

// ---------------------------------------------------------------------------
// Problem constants
// ---------------------------------------------------------------------------
#define BB 8
#define TT 1024
#define DD 1024
#define VV 1024
#define HH 512
#define GG 2048            // 4*H
#define MM (BB * TT)       // 8192 rows

// ---------------------------------------------------------------------------
// Tiny zero-fill (64B zero pad used for conv boundary rows)
// ---------------------------------------------------------------------------
__global__ void k_zero(float* p) { p[threadIdx.x] = 0.0f; }

// ---------------------------------------------------------------------------
// Elementwise cast f32 -> bf16
// ---------------------------------------------------------------------------
__global__ void k_cast(const float* __restrict__ s, __bf16* __restrict__ d, long n) {
  long i = (long)blockIdx.x * blockDim.x + threadIdx.x;
  if (i < n) d[i] = (__bf16)s[i];
}

// ---------------------------------------------------------------------------
// Transpose + cast: src f32 [R,C] -> dst bf16 [C,R]
// ---------------------------------------------------------------------------
__global__ void k_tcast(const float* __restrict__ src, __bf16* __restrict__ dst,
                        int R, int C) {
  __shared__ float t[32][33];
  int c0 = blockIdx.x * 32, r0 = blockIdx.y * 32;
#pragma unroll
  for (int i = threadIdx.y; i < 32; i += 8)
    t[i][threadIdx.x] = src[(size_t)(r0 + i) * C + c0 + threadIdx.x];
  __syncthreads();
#pragma unroll
  for (int i = threadIdx.y; i < 32; i += 8)
    dst[(size_t)(c0 + i) * R + r0 + threadIdx.x] = (__bf16)t[threadIdx.x][i];
}

// ---------------------------------------------------------------------------
// Generic WMMA GEMM / implicit width-5 conv:
//   out[m,n] = act( sum_tap sum_k A[rowshift(m,tap), k] * Bt[tap, n, k] + bias[n] )
// Block: 256 threads (8 waves). Block tile 128x128, wave tile 32x64.
// Double-buffered LDS tiles; staging via global_load_async_to_lds_b128
// software-pipelined one tile ahead (s_wait_asynccnt 0x4 in steady state).
// ---------------------------------------------------------------------------
__global__ __launch_bounds__(256) void k_gemm(
    const __bf16* __restrict__ A, const __bf16* __restrict__ Bt,
    const float* __restrict__ bias, const __bf16* __restrict__ zeros,
    float* __restrict__ outF, __bf16* __restrict__ outB,
    int M, int N, int K, int taps, int Tlen, int act) {
  __shared__ __bf16 As[2][128 * 40];
  __shared__ __bf16 Bs[2][128 * 40];

  int tid = threadIdx.x;
  int wave = tid >> 5, lane = tid & 31;
  int wm = wave & 3, wn = wave >> 2;          // 4 M-strips x 2 N-strips
  int m0 = blockIdx.y * 128, n0g = blockIdx.x * 128;
  int l = lane & 15, hsel = lane >> 4;

  v8f acc[2][4];
#pragma unroll
  for (int mi = 0; mi < 2; ++mi)
#pragma unroll
    for (int ni = 0; ni < 4; ++ni) acc[mi][ni] = zero_v8f();

  // ---- tile-fill assignment: each thread owns one 32B row segment ----
  int row = tid >> 1, half = tid & 1;
  unsigned ldsA[2] = {lds_off(&As[0][row * 40 + half * 16]),
                      lds_off(&As[1][row * 40 + half * 16])};
  unsigned ldsB[2] = {lds_off(&Bs[0][row * 40 + half * 16]),
                      lds_off(&Bs[1][row * 40 + half * 16])};
  int gr = m0 + row;
  int bbatch = gr / Tlen;
  int trow = gr - bbatch * Tlen;
  const __bf16* Abase = A + (size_t)bbatch * Tlen * K + half * 16;
  const __bf16* Bbase = Bt + ((size_t)n0g + row) * K + half * 16;

  auto issue = [&](int tap, int k0, int buf) {
    int shift = (taps > 1) ? (tap - 2) : 0;
    int ts = trow + shift;
    const __bf16* ap =
        (ts >= 0 && ts < Tlen) ? (Abase + (size_t)ts * K + k0) : zeros;
    async_ld_b128(ldsA[buf], ap);
    async_ld_b128(ldsA[buf] + 16, ap + 8);
    const __bf16* bp = Bbase + (size_t)tap * N * K + k0;
    async_ld_b128(ldsB[buf], bp);
    async_ld_b128(ldsB[buf] + 16, bp + 8);
  };

  int kTiles = K / 32;
  int nIter = taps * kTiles;
  int tap = 0, k0 = 0, buf = 0;
  issue(0, 0, 0);

  for (int it = 0; it < nIter; ++it) {
    // next tile coordinates
    int nk0 = k0 + 32, ntap = tap;
    if (nk0 == K) { nk0 = 0; ++ntap; }
    // pipeline: issue tile i+1 into the other buffer, then wait for tile i
    if (it + 1 < nIter) {
      issue(ntap, nk0, buf ^ 1);
      wait_async4();
    } else {
      wait_async0();
    }
    __syncthreads();  // tile i visible to all waves

    // ---- fragments + WMMA from LDS buffer `buf` ----
    const __bf16* AsB = As[buf];
    const __bf16* BsB = Bs[buf];
    v16bf af[2];
#pragma unroll
    for (int mi = 0; mi < 2; ++mi) {
      int r = wm * 32 + mi * 16 + l;
      const v8bf* p0 = (const v8bf*)&AsB[r * 40 + hsel * 8];
      const v8bf* p1 = (const v8bf*)&AsB[r * 40 + 16 + hsel * 8];
#pragma unroll
      for (int i = 0; i < 8; ++i) {
        af[mi][i] = (*p0)[i];
        af[mi][i + 8] = (*p1)[i];
      }
    }
#pragma unroll
    for (int ni = 0; ni < 4; ++ni) {
      int rb = wn * 64 + ni * 16 + l;
      const v8bf* q0 = (const v8bf*)&BsB[rb * 40 + hsel * 8];
      const v8bf* q1 = (const v8bf*)&BsB[rb * 40 + 16 + hsel * 8];
      v16bf bfr;
#pragma unroll
      for (int i = 0; i < 8; ++i) {
        bfr[i] = (*q0)[i];
        bfr[i + 8] = (*q1)[i];
      }
      acc[0][ni] = wmma_bf16(af[0], bfr, acc[0][ni]);
      acc[1][ni] = wmma_bf16(af[1], bfr, acc[1][ni]);
    }
    __syncthreads();  // buffer free for the issue two iterations ahead

    tap = ntap;
    k0 = nk0;
    buf ^= 1;
  }

  // ---- epilogue: bias + activation + store ----
#pragma unroll
  for (int mi = 0; mi < 2; ++mi)
#pragma unroll
    for (int ni = 0; ni < 4; ++ni) {
#pragma unroll
      for (int r = 0; r < 8; ++r) {
        int m = m0 + wm * 32 + mi * 16 + r + 8 * hsel;
        int n = n0g + wn * 64 + ni * 16 + l;
        float v = acc[mi][ni][r] + bias[n];
        if (act == 1) v = fmaxf(v, 0.0f);
        size_t idx = (size_t)m * N + n;
        if (outF) outF[idx] = v;
        if (outB) outB[idx] = (__bf16)v;
      }
    }
}

// ---------------------------------------------------------------------------
// LSTM recurrence (one direction per block; block = 1024 threads = 32 waves).
// Each wave owns a 16-wide slab of H; per step: 4 gate tiles x 16 K-steps of
// v_wmma_f32_16x16x32_bf16 with h_prev (bf16, LDS) x whT (bf16, L2-resident).
// xg = precomputed x-projection (f32, [B,T,4H]); hs out f32 [B,T,H].
// ---------------------------------------------------------------------------
__global__ __launch_bounds__(1024) void k_lstm(
    const float* __restrict__ xg_f, const float* __restrict__ xg_b,
    const __bf16* __restrict__ whT_f, const __bf16* __restrict__ whT_b,
    float* __restrict__ hs_f, float* __restrict__ hs_b) {
  const int HSTR = 520;  // padded LDS row stride (halves)
  int dir = blockIdx.x;
  const float* xg = dir ? xg_b : xg_f;
  const __bf16* whT = dir ? whT_b : whT_f;
  float* hs = dir ? hs_b : hs_f;

  __shared__ __bf16 Hs[16 * HSTR];

  int tid = threadIdx.x;
  for (int i = tid; i < 16 * HSTR; i += 1024) Hs[i] = (__bf16)0.0f;
  __syncthreads();

  int wave = tid >> 5, lane = tid & 31;
  int n0 = wave * 16;
  int l = lane & 15, hsel = lane >> 4;

  v8f c = zero_v8f();

  for (int t = 0; t < TT; ++t) {
    int tt = dir ? (TT - 1 - t) : t;

    // seed accumulators with xg (gate order: i, f, g, o along 4H)
    v8f acc[4];
#pragma unroll
    for (int g = 0; g < 4; ++g) {
#pragma unroll
      for (int r = 0; r < 8; ++r) {
        float v = xg[((size_t)r * TT + tt) * GG + g * HH + n0 + l];
        acc[g][r] = (hsel == 0) ? v : 0.0f;
      }
    }

    // h_prev @ whT over K=512
    for (int k0 = 0; k0 < HH; k0 += 32) {
      v16bf a;
      {
        const v8bf* p0 = (const v8bf*)&Hs[l * HSTR + k0 + hsel * 8];
        const v8bf* p1 = (const v8bf*)&Hs[l * HSTR + k0 + 16 + hsel * 8];
#pragma unroll
        for (int i = 0; i < 8; ++i) {
          a[i] = (*p0)[i];
          a[i + 8] = (*p1)[i];
        }
      }
#pragma unroll
      for (int g = 0; g < 4; ++g) {
        const __bf16* bp =
            &whT[((size_t)(g * HH + n0 + l)) * HH + k0 + hsel * 8];
        if (k0 + 32 < HH) __builtin_prefetch(bp + 32, 0, 1);
        const v8bf* pb = (const v8bf*)bp;
        const v8bf* pb1 = (const v8bf*)(bp + 16);
        v16bf b;
#pragma unroll
        for (int i = 0; i < 8; ++i) {
          b[i] = pb[0][i];
          b[i + 8] = pb1[0][i];
        }
        acc[g] = wmma_bf16(a, b, acc[g]);
      }
    }

    __syncthreads();  // all waves done reading h_prev from LDS

    if (hsel == 0) {  // lanes 0..15 hold the valid batch rows (M=0..7)
#pragma unroll
      for (int r = 0; r < 8; ++r) {
        float iv = sigmoidf_(acc[0][r]);
        float fv = sigmoidf_(acc[1][r]);
        float gv = tanhf(acc[2][r]);
        float ov = sigmoidf_(acc[3][r]);
        float cn = fv * c[r] + iv * gv;
        c[r] = cn;
        float h = ov * tanhf(cn);
        Hs[r * HSTR + n0 + l] = (__bf16)h;
        hs[((size_t)r * TT + tt) * HH + n0 + l] = h;
      }
    }
    __syncthreads();  // new h visible before next step
  }
}

// ---------------------------------------------------------------------------
// Concat fwd/bwd hidden states -> bf16 [M, 2H]
// ---------------------------------------------------------------------------
__global__ void k_concat(const float* __restrict__ hsf,
                         const float* __restrict__ hsb,
                         __bf16* __restrict__ cat, long n) {
  long i = (long)blockIdx.x * blockDim.x + threadIdx.x;
  if (i >= n) return;
  long r = i / DD;
  int j = (int)(i - r * DD);
  float v = (j < HH) ? hsf[r * HH + j] : hsb[r * HH + (j - HH)];
  cat[i] = (__bf16)v;
}

// ---------------------------------------------------------------------------
// BatchNorm column stats over M rows: mean + rsqrt(var + eps)
// ---------------------------------------------------------------------------
__global__ __launch_bounds__(256) void k_stats(const float* __restrict__ x,
                                               float* __restrict__ mean,
                                               float* __restrict__ rstd,
                                               int Mrows, int C) {
  int cch = blockIdx.x;
  float s = 0.0f, ss = 0.0f;
  for (int r = threadIdx.x; r < Mrows; r += 256) {
    float v = x[(size_t)r * C + cch];
    s += v;
    ss += v * v;
  }
  __shared__ float sm[256], sm2[256];
  sm[threadIdx.x] = s;
  sm2[threadIdx.x] = ss;
  __syncthreads();
  for (int w = 128; w > 0; w >>= 1) {
    if (threadIdx.x < w) {
      sm[threadIdx.x] += sm[threadIdx.x + w];
      sm2[threadIdx.x] += sm2[threadIdx.x + w];
    }
    __syncthreads();
  }
  if (threadIdx.x == 0) {
    float mu = sm[0] / (float)Mrows;
    float var = sm2[0] / (float)Mrows - mu * mu;
    mean[cch] = mu;
    rstd[cch] = rsqrtf(var + 1e-5f);
  }
}

// ---------------------------------------------------------------------------
// Fused BN normalize + affine + tanh -> bf16
// ---------------------------------------------------------------------------
__global__ void k_bntanh(const float* __restrict__ x,
                         const float* __restrict__ mean,
                         const float* __restrict__ rstd,
                         const float* __restrict__ g,
                         const float* __restrict__ be,
                         __bf16* __restrict__ out, int C, long n) {
  long i = (long)blockIdx.x * blockDim.x + threadIdx.x;
  if (i >= n) return;
  int cch = (int)(i % C);
  float v = (x[i] - mean[cch]) * rstd[cch] * g[cch] + be[cch];
  out[i] = (__bf16)tanhf(v);
}

// ---------------------------------------------------------------------------
// Host orchestration
// ---------------------------------------------------------------------------
extern "C" void kernel_launch(void* const* d_in, const int* in_sizes, int n_in,
                              void* d_out, int out_size, void* d_ws,
                              size_t ws_size, hipStream_t stream) {
  (void)in_sizes; (void)n_in; (void)out_size; (void)ws_size;

  const float* x    = (const float*)d_in[0];
  const float* w1   = (const float*)d_in[1];
  const float* b1   = (const float*)d_in[2];
  const float* w2   = (const float*)d_in[3];
  const float* b2   = (const float*)d_in[4];
  const float* wx_f = (const float*)d_in[5];
  const float* wh_f = (const float*)d_in[6];
  const float* bl_f = (const float*)d_in[7];
  const float* wx_b = (const float*)d_in[8];
  const float* wh_b = (const float*)d_in[9];
  const float* bl_b = (const float*)d_in[10];
  const float* ck1  = (const float*)d_in[11];
  const float* cb1  = (const float*)d_in[12];
  const float* g1   = (const float*)d_in[13];
  const float* be1  = (const float*)d_in[14];
  const float* ck2  = (const float*)d_in[15];
  const float* cb2  = (const float*)d_in[16];
  const float* g2   = (const float*)d_in[17];
  const float* be2  = (const float*)d_in[18];
  const float* ck3  = (const float*)d_in[19];
  const float* cb3  = (const float*)d_in[20];
  float* out = (float*)d_out;

  char* ws = (char*)d_ws;
  size_t off = 0;
  auto alloc = [&](size_t bytes) {
    size_t o = off;
    off += (bytes + 255) & ~(size_t)255;
    return o;
  };

  // zero pad for conv boundary rows (64B used; alloc 256B)
  float* zpad = (float*)(ws + alloc(256));
  // bf16 weight buffers (persist for whole launch)
  __bf16* w1t  = (__bf16*)(ws + alloc((size_t)DD * DD * 2));
  __bf16* w2t  = (__bf16*)(ws + alloc((size_t)DD * DD * 2));
  __bf16* wxft = (__bf16*)(ws + alloc((size_t)GG * DD * 2));
  __bf16* wxbt = (__bf16*)(ws + alloc((size_t)GG * DD * 2));
  __bf16* whft = (__bf16*)(ws + alloc((size_t)GG * HH * 2));
  __bf16* whbt = (__bf16*)(ws + alloc((size_t)GG * HH * 2));
  __bf16* ck1t = (__bf16*)(ws + alloc((size_t)5 * DD * DD * 2));
  __bf16* ck2t = (__bf16*)(ws + alloc((size_t)5 * DD * DD * 2));
  __bf16* ck3t = (__bf16*)(ws + alloc((size_t)5 * VV * DD * 2));
  // activations
  __bf16* x_bf  = (__bf16*)(ws + alloc((size_t)MM * DD * 2));
  __bf16* h1_bf = (__bf16*)(ws + alloc((size_t)MM * DD * 2));
  __bf16* h2_bf = (__bf16*)(ws + alloc((size_t)MM * DD * 2));
  float*  hs_f  = (float*) (ws + alloc((size_t)MM * HH * 4));
  float*  hs_b  = (float*) (ws + alloc((size_t)MM * HH * 4));
  __bf16* catb  = (__bf16*)(ws + alloc((size_t)MM * DD * 2));
  float*  mean1 = (float*) (ws + alloc((size_t)DD * 4));
  float*  rstd1 = (float*) (ws + alloc((size_t)DD * 4));
  float*  mean2 = (float*) (ws + alloc((size_t)DD * 4));
  float*  rstd2 = (float*) (ws + alloc((size_t)DD * 4));
  // big overlay region: xg (LSTM phase) then conv buffers (conv phase)
  char* big = ws + alloc((size_t)128 * 1024 * 1024);
  float*  xg_f = (float*)(big);
  float*  xg_b = (float*)(big + (size_t)MM * GG * 4);
  float*  c1f  = (float*)(big);                                   // 32MB
  __bf16* c1b  = (__bf16*)(big + (size_t)MM * DD * 4);            // 16MB
  float*  c2f  = (float*)(big + (size_t)MM * DD * 6);             // 32MB
  __bf16* c2b  = (__bf16*)(big + (size_t)MM * DD * 10);           // 16MB
  const __bf16* zpad_bf = (const __bf16*)zpad;

  dim3 tb(32, 8);

  // ---- zero pad init ----
  k_zero<<<1, 32, 0, stream>>>(zpad);

  // ---- weight prep: transpose + cast to bf16 ----
  k_tcast<<<dim3(DD / 32, DD / 32), tb, 0, stream>>>(w1, w1t, DD, DD);
  k_tcast<<<dim3(DD / 32, DD / 32), tb, 0, stream>>>(w2, w2t, DD, DD);
  k_tcast<<<dim3(GG / 32, DD / 32), tb, 0, stream>>>(wx_f, wxft, DD, GG);
  k_tcast<<<dim3(GG / 32, DD / 32), tb, 0, stream>>>(wx_b, wxbt, DD, GG);
  k_tcast<<<dim3(GG / 32, HH / 32), tb, 0, stream>>>(wh_f, whft, HH, GG);
  k_tcast<<<dim3(GG / 32, HH / 32), tb, 0, stream>>>(wh_b, whbt, HH, GG);
  for (int tap = 0; tap < 5; ++tap) {
    k_tcast<<<dim3(DD / 32, DD / 32), tb, 0, stream>>>(
        ck1 + (size_t)tap * DD * DD, ck1t + (size_t)tap * DD * DD, DD, DD);
    k_tcast<<<dim3(DD / 32, DD / 32), tb, 0, stream>>>(
        ck2 + (size_t)tap * DD * DD, ck2t + (size_t)tap * DD * DD, DD, DD);
    k_tcast<<<dim3(VV / 32, DD / 32), tb, 0, stream>>>(
        ck3 + (size_t)tap * DD * VV, ck3t + (size_t)tap * VV * DD, DD, VV);
  }

  // ---- cast input ----
  {
    long n = (long)MM * DD;
    k_cast<<<(unsigned)((n + 255) / 256), 256, 0, stream>>>(x, x_bf, n);
  }

  // ---- dense1 (ReLU) and dense2 ----
  k_gemm<<<dim3(DD / 128, MM / 128), 256, 0, stream>>>(
      x_bf, w1t, b1, zpad_bf, nullptr, h1_bf, MM, DD, DD, 1, TT, 1);
  k_gemm<<<dim3(DD / 128, MM / 128), 256, 0, stream>>>(
      h1_bf, w2t, b2, zpad_bf, nullptr, h2_bf, MM, DD, DD, 1, TT, 0);

  // ---- LSTM input projections: xg = h2 @ wx + bl ----
  k_gemm<<<dim3(GG / 128, MM / 128), 256, 0, stream>>>(
      h2_bf, wxft, bl_f, zpad_bf, xg_f, nullptr, MM, GG, DD, 1, TT, 0);
  k_gemm<<<dim3(GG / 128, MM / 128), 256, 0, stream>>>(
      h2_bf, wxbt, bl_b, zpad_bf, xg_b, nullptr, MM, GG, DD, 1, TT, 0);

  // ---- bidirectional LSTM recurrence ----
  k_lstm<<<2, 1024, 0, stream>>>(xg_f, xg_b, whft, whbt, hs_f, hs_b);

  // ---- concat ----
  {
    long n = (long)MM * DD;
    k_concat<<<(unsigned)((n + 255) / 256), 256, 0, stream>>>(hs_f, hs_b, catb, n);
  }

  // ---- conv1 (implicit GEMM, 5 taps) + BN + tanh ----
  k_gemm<<<dim3(DD / 128, MM / 128), 256, 0, stream>>>(
      catb, ck1t, cb1, zpad_bf, c1f, nullptr, MM, DD, DD, 5, TT, 0);
  k_stats<<<DD, 256, 0, stream>>>(c1f, mean1, rstd1, MM, DD);
  {
    long n = (long)MM * DD;
    k_bntanh<<<(unsigned)((n + 255) / 256), 256, 0, stream>>>(
        c1f, mean1, rstd1, g1, be1, c1b, DD, n);
  }

  // ---- conv2 + BN + tanh ----
  k_gemm<<<dim3(DD / 128, MM / 128), 256, 0, stream>>>(
      c1b, ck2t, cb2, zpad_bf, c2f, nullptr, MM, DD, DD, 5, TT, 0);
  k_stats<<<DD, 256, 0, stream>>>(c2f, mean2, rstd2, MM, DD);
  {
    long n = (long)MM * DD;
    k_bntanh<<<(unsigned)((n + 255) / 256), 256, 0, stream>>>(
        c2f, mean2, rstd2, g2, be2, c2b, DD, n);
  }

  // ---- conv3 -> output (f32) ----
  k_gemm<<<dim3(VV / 128, MM / 128), 256, 0, stream>>>(
      c2b, ck3t, cb3, zpad_bf, out, nullptr, MM, VV, DD, 5, TT, 0);
}
